// GATv2MultiTask_70720931496091
// MI455X (gfx1250) — matrix operand order repaired and checked by reading, not compiled, not run on hardware
//
#include <hip/hip_runtime.h>

// ---------------------------------------------------------------------------
// GATv2 multi-task forward for MI455X (gfx1250, wave32, WMMA).
// Dense node transforms use v_wmma_f32_16x16x32_bf16 (bf16 in / f32 acc).
// Edge phase (segment softmax + scatter) is gather/scatter bound -> streaming
// passes with float atomics. Pooling is deterministic (batch is sorted).
// ---------------------------------------------------------------------------

typedef __attribute__((ext_vector_type(16))) __bf16 v16bf;
typedef __attribute__((ext_vector_type(8)))  __bf16 v8bf;
typedef __attribute__((ext_vector_type(8)))  float  v8f;

namespace gat {

constexpr int N  = 50000;     // nodes  (== 3125 * 16, exact WMMA M tiling)
constexpr int E  = 500000;    // edges
constexpr int E2 = E + N;     // edges + self loops
constexpr int B  = 50;        // graphs
constexpr int H  = 4;         // heads
constexpr int D  = 64;        // dim per head
constexpr int C  = 256;       // H*D channels
constexpr float NEG = 0.2f;   // leaky-relu slope

__device__ __forceinline__ void edge_sd(const int* __restrict__ ei, int e,
                                        int& s, int& d) {
  if (e < E) { s = ei[e]; d = ei[E + e]; }
  else       { s = e - E; d = e - E; }          // self loop
}

// Sign-correct float atomic max via integer atomics (deterministic).
__device__ __forceinline__ void atomicMaxF(float* addr, float val) {
  if (val >= 0.0f) {
    atomicMax(reinterpret_cast<int*>(addr), __float_as_int(val));
  } else {
    atomicMin(reinterpret_cast<unsigned int*>(addr), __float_as_uint(val));
  }
}

// ------------------------------- utility -----------------------------------

__global__ void k_fill_f32(float* __restrict__ p, float v, int n) {
  int i = blockIdx.x * blockDim.x + threadIdx.x;
  if (i < n) p[i] = v;
}

__global__ __launch_bounds__(256) void k_ea_part(const float* __restrict__ ea,
                                                 float* __restrict__ part) {
  __shared__ float red[256];
  int j = threadIdx.x;
  float acc = 0.0f;
  for (int i = blockIdx.x * 256 + j; i < E; i += 256 * 256) acc += ea[i];
  red[j] = acc; __syncthreads();
  for (int s = 128; s > 0; s >>= 1) { if (j < s) red[j] += red[j + s]; __syncthreads(); }
  if (j == 0) part[blockIdx.x] = red[0];
}

__global__ __launch_bounds__(256) void k_ea_fin(const float* __restrict__ part,
                                                float* __restrict__ mean) {
  __shared__ float red[256];
  int j = threadIdx.x;
  red[j] = part[j]; __syncthreads();
  for (int s = 128; s > 0; s >>= 1) { if (j < s) red[j] += red[j + s]; __syncthreads(); }
  if (j == 0) mean[0] = red[0] / (float)E;
}

// encoder: h0 = relu(x @ enc_w + enc_b), stored bf16 for the layer-1 WMMA GEMM
__global__ void k_encoder(const float* __restrict__ x,
                          const float* __restrict__ w,
                          const float* __restrict__ b,
                          __bf16* __restrict__ h0) {
  int tid = blockIdx.x * blockDim.x + threadIdx.x;
  if (tid >= N * 64) return;
  int n = tid >> 6, j = tid & 63;
  const float* xr = x + n * 8;
  float acc = b[j];
#pragma unroll
  for (int k = 0; k < 8; ++k) acc += xr[k] * w[k * 64 + j];
  h0[tid] = (__bf16)fmaxf(acc, 0.0f);
}

// W[K,256] f32 (row major) -> W^T[256][K] bf16 (n-major, K contiguous)
__global__ void k_twt(const float* __restrict__ w, __bf16* __restrict__ wt, int K) {
  int tid = blockIdx.x * blockDim.x + threadIdx.x;
  if (tid >= K * 256) return;
  int n = tid / K, k = tid - n * K;
  wt[tid] = (__bf16)w[k * 256 + n];
}

// ------------------------------ WMMA GEMM ----------------------------------
// out{L,R}[N x 256] = A[N x K] (bf16) @ W{l,r}[K x 256] + bias{l,r}
// One 16x16 tile per wave-accumulator. Block = 8 waves; each wave owns
// n-tiles {w, w+8} for BOTH weight matrices (A fragment reused 4x).
// A/B fragments use contiguous k (k = 16*kh + s), not the ISA table order; f32 C/D per CDNA5 ISA 7.12.2.
template <int K>
__global__ __launch_bounds__(256)
void k_gemm_dual(const __bf16* __restrict__ A,
                 const __bf16* __restrict__ WTl,
                 const __bf16* __restrict__ WTr,
                 const float* __restrict__ bl,
                 const float* __restrict__ br,
                 float* __restrict__ outL,
                 float* __restrict__ outR) {
  const int mtile = blockIdx.x;               // 3125 tiles of 16 rows
  const int wave  = threadIdx.x >> 5;         // 0..7
  const int lane  = threadIdx.x & 31;
  const int lm    = lane & 15;                // row/col within tile
  const int kh    = lane >> 4;                // K-half select
  const int row   = mtile * 16 + lm;
  const int col0  = (wave    ) * 16 + lm;
  const int col1  = (wave + 8) * 16 + lm;

  v8f aL0 = {}; v8f aL1 = {}; v8f aR0 = {}; v8f aR1 = {};

  const __bf16* arow = A + (size_t)row * K;
  const __bf16* bl0  = WTl + (size_t)col0 * K;
  const __bf16* bl1  = WTl + (size_t)col1 * K;
  const __bf16* br0  = WTr + (size_t)col0 * K;
  const __bf16* br1  = WTr + (size_t)col1 * K;

#pragma unroll
  for (int k0 = 0; k0 < K; k0 += 32) {
    // A fragment: contiguous 16 K-values starting at k0 + kh*16
    v16bf av = *reinterpret_cast<const v16bf*>(arow + k0 + kh * 16);

    // B fragment: contiguous 16 K-values starting at k0 + kh*16 (W^T layout)
    v16bf bL0v = *reinterpret_cast<const v16bf*>(bl0 + k0 + kh * 16);
    v16bf bL1v = *reinterpret_cast<const v16bf*>(bl1 + k0 + kh * 16);
    v16bf bR0v = *reinterpret_cast<const v16bf*>(br0 + k0 + kh * 16);
    v16bf bR1v = *reinterpret_cast<const v16bf*>(br1 + k0 + kh * 16);

    aL0 = __builtin_amdgcn_wmma_f32_16x16x32_bf16(false, av, false, bL0v, (short)0, aL0, false, false);
    aL1 = __builtin_amdgcn_wmma_f32_16x16x32_bf16(false, av, false, bL1v, (short)0, aL1, false, false);
    aR0 = __builtin_amdgcn_wmma_f32_16x16x32_bf16(false, av, false, bR0v, (short)0, aR0, false, false);
    aR1 = __builtin_amdgcn_wmma_f32_16x16x32_bf16(false, av, false, bR1v, (short)0, aR1, false, false);
  }

  // C/D layout: lane 0-15 -> M = j (VGPR j), N = lane; lane 16-31 -> M = 8+j.
  const float bL0s = bl[col0], bL1s = bl[col1];
  const float bR0s = br[col0], bR1s = br[col1];
#pragma unroll
  for (int j = 0; j < 8; ++j) {
    size_t r = (size_t)(mtile * 16 + kh * 8 + j) * C;
    outL[r + col0] = aL0[j] + bL0s;
    outL[r + col1] = aL1[j] + bL1s;
    outR[r + col0] = aR0[j] + bR0s;
    outR[r + col1] = aR1[j] + bR1s;
  }
}

// ------------------------------ edge phase ---------------------------------

// pass 1: per (edge, head) attention logit + segment max into lmax[dst][h]
__global__ void k_edge_logits(const int* __restrict__ ei,
                              const float* __restrict__ ea,
                              const float* __restrict__ eamean,
                              const float* __restrict__ xl,
                              const float* __restrict__ xr,
                              const float* __restrict__ we,
                              const float* __restrict__ att,
                              float* __restrict__ logits,
                              float* __restrict__ lmax) {
  int tid = blockIdx.x * blockDim.x + threadIdx.x;
  if (tid >= E2 * H) return;
  int e = tid >> 2, h = tid & 3;
  int s, d; edge_sd(ei, e, s, d);
  float eav = (e < E) ? ea[e] : eamean[0];
  const float* pl = xl + (size_t)s * C + h * D;
  const float* pr = xr + (size_t)d * C + h * D;
  const float* pw = we + h * D;
  const float* pa = att + h * D;
  float acc = 0.0f;
#pragma unroll 4
  for (int i = 0; i < D; i += 4) {
    float4 a = *reinterpret_cast<const float4*>(pl + i);
    float4 b = *reinterpret_cast<const float4*>(pr + i);
    float4 w = *reinterpret_cast<const float4*>(pw + i);
    float4 t = *reinterpret_cast<const float4*>(pa + i);
    float m0 = a.x + b.x + eav * w.x; m0 = m0 > 0.0f ? m0 : m0 * NEG;
    float m1 = a.y + b.y + eav * w.y; m1 = m1 > 0.0f ? m1 : m1 * NEG;
    float m2 = a.z + b.z + eav * w.z; m2 = m2 > 0.0f ? m2 : m2 * NEG;
    float m3 = a.w + b.w + eav * w.w; m3 = m3 > 0.0f ? m3 : m3 * NEG;
    acc += m0 * t.x + m1 * t.y + m2 * t.z + m3 * t.w;
  }
  logits[tid] = acc;
  atomicMaxF(lmax + (size_t)d * H + h, acc);
}

// pass 2: ex = exp(logit - lmax[dst]); den[dst][h] += ex (ex overwrites logits)
__global__ void k_edge_exp(const int* __restrict__ ei,
                           float* __restrict__ logits,
                           const float* __restrict__ lmax,
                           float* __restrict__ den) {
  int tid = blockIdx.x * blockDim.x + threadIdx.x;
  if (tid >= E2 * H) return;
  int e = tid >> 2, h = tid & 3;
  int s, d; edge_sd(ei, e, s, d);
  float ex = __expf(logits[tid] - lmax[(size_t)d * H + h]);
  logits[tid] = ex;
  atomicAdd(den + (size_t)d * H + h, ex);
}

// pass 3: agg[dst][c] += xl[src][c] * (ex / den[dst][h]); 64 lanes/edge, 4 c each
__global__ void k_edge_agg(const int* __restrict__ ei,
                           const float* __restrict__ ex,
                           const float* __restrict__ den,
                           const float* __restrict__ xl,
                           float* __restrict__ agg) {
  int tid = blockIdx.x * blockDim.x + threadIdx.x;
  if (tid >= E2 * 64) return;
  int e  = tid >> 6;
  int c0 = (tid & 63) * 4;
  int h  = c0 >> 6;
  int s, d; edge_sd(ei, e, s, d);
  float alpha = ex[(size_t)e * H + h] / den[(size_t)d * H + h];
  float4 v = *reinterpret_cast<const float4*>(xl + (size_t)s * C + c0);
  float* pd = agg + (size_t)d * C + c0;
  atomicAdd(pd + 0, v.x * alpha);
  atomicAdd(pd + 1, v.y * alpha);
  atomicAdd(pd + 2, v.z * alpha);
  atomicAdd(pd + 3, v.w * alpha);
}

// h = relu(agg + bias): optional f32 (in place ok) and/or bf16 output
__global__ void k_postnode(const float* __restrict__ agg,
                           const float* __restrict__ bias,
                           float* __restrict__ of,
                           __bf16* __restrict__ ob) {
  int tid = blockIdx.x * blockDim.x + threadIdx.x;
  if (tid >= N * C) return;
  float v = fmaxf(agg[tid] + bias[tid & 255], 0.0f);
  if (of) of[tid] = v;
  if (ob) ob[tid] = (__bf16)v;
}

// ------------------------------- pooling -----------------------------------

__global__ void k_bounds_init(int* __restrict__ gs, int* __restrict__ ge) {
  int b = threadIdx.x;
  if (b < B) { gs[b] = 0; ge[b] = 0; }
}

__global__ void k_bounds(const int* __restrict__ batch,
                         int* __restrict__ gs, int* __restrict__ ge) {
  int n = blockIdx.x * blockDim.x + threadIdx.x;
  if (n >= N) return;
  int b = batch[n];
  if (n == 0 || batch[n - 1] != b) gs[b] = n;
  if (n == N - 1 || batch[n + 1] != b) ge[b] = n + 1;
}

__global__ __launch_bounds__(256)
void k_pool(const float* __restrict__ h2, const int* __restrict__ gs,
            const int* __restrict__ ge, float* __restrict__ pooled) {
  int b = blockIdx.x, c = threadIdx.x;
  int s = gs[b], e = ge[b];
  float acc = 0.0f;
  for (int n = s; n < e; ++n) acc += h2[(size_t)n * C + c];
  int cnt = e - s;
  pooled[b * C + c] = acc / (float)(cnt > 0 ? cnt : 1);
}

// ------------------------------- MLP head ----------------------------------

__global__ __launch_bounds__(128)
void k_head(const float* __restrict__ pooled,
            const float* __restrict__ p1w, const float* __restrict__ p1b,
            const float* __restrict__ lng, const float* __restrict__ lnb,
            const float* __restrict__ p2w, const float* __restrict__ p2b,
            const float* __restrict__ hcw, const float* __restrict__ hcb,
            const float* __restrict__ hfw, const float* __restrict__ hfb,
            float* __restrict__ out) {
  __shared__ float st[128], sz[64], red[128];
  int b = blockIdx.x, j = threadIdx.x;
  const float* pb = pooled + b * C;

  float t = p1b[j];
  for (int k = 0; k < C; ++k) t += pb[k] * p1w[k * 128 + j];

  red[j] = t; __syncthreads();
  for (int s = 64; s > 0; s >>= 1) { if (j < s) red[j] += red[j + s]; __syncthreads(); }
  float mu = red[0] / 128.0f; __syncthreads();
  float dm = t - mu;
  red[j] = dm * dm; __syncthreads();
  for (int s = 64; s > 0; s >>= 1) { if (j < s) red[j] += red[j + s]; __syncthreads(); }
  float var = red[0] / 128.0f;

  float tn = dm * rsqrtf(var + 1e-5f) * lng[j] + lnb[j];
  st[j] = fmaxf(tn, 0.0f);
  __syncthreads();

  if (j < 64) {
    float z = p2b[j];
    for (int k = 0; k < 128; ++k) z += st[k] * p2w[k * 64 + j];
    z = fmaxf(z, 0.0f);
    sz[j] = z;
    out[500 + b * 64 + j] = z;                 // z block
  }
  __syncthreads();
  if (j == 0) {
    float hc = hcb[0];
    for (int k = 0; k < 64; ++k) hc += sz[k] * hcw[k];
    out[b] = hc;                               // cooling logit
  }
  if (j < 9) {
    float hf = hfb[j];
    for (int k = 0; k < 64; ++k) hf += sz[k] * hfw[k * 9 + j];
    out[50 + b * 9 + j] = hf;                  // fatigue logits
  }
}

}  // namespace gat

// ---------------------------------------------------------------------------

extern "C" void kernel_launch(void* const* d_in, const int* in_sizes, int n_in,
                              void* d_out, int out_size, void* d_ws, size_t ws_size,
                              hipStream_t stream) {
  using namespace gat;
  (void)in_sizes; (void)n_in; (void)out_size; (void)ws_size;

  const float* x      = (const float*)d_in[0];
  const int*   ei     = (const int*)  d_in[1];
  const float* ea     = (const float*)d_in[2];
  const int*   batch  = (const int*)  d_in[3];
  const float* enc_w  = (const float*)d_in[4];  const float* enc_b  = (const float*)d_in[5];
  const float* g1_wl  = (const float*)d_in[6];  const float* g1_bl  = (const float*)d_in[7];
  const float* g1_wr  = (const float*)d_in[8];  const float* g1_br  = (const float*)d_in[9];
  const float* g1_we  = (const float*)d_in[10]; const float* g1_att = (const float*)d_in[11];
  const float* g1_bias= (const float*)d_in[12];
  const float* g2_wl  = (const float*)d_in[13]; const float* g2_bl  = (const float*)d_in[14];
  const float* g2_wr  = (const float*)d_in[15]; const float* g2_br  = (const float*)d_in[16];
  const float* g2_we  = (const float*)d_in[17]; const float* g2_att = (const float*)d_in[18];
  const float* g2_bias= (const float*)d_in[19];
  const float* p1_w = (const float*)d_in[20]; const float* p1_b = (const float*)d_in[21];
  const float* ln_g = (const float*)d_in[22]; const float* ln_b = (const float*)d_in[23];
  const float* p2_w = (const float*)d_in[24]; const float* p2_b = (const float*)d_in[25];
  const float* hc_w = (const float*)d_in[26]; const float* hc_b = (const float*)d_in[27];
  const float* hf_w = (const float*)d_in[28]; const float* hf_b = (const float*)d_in[29];
  float* out = (float*)d_out;

  // ---- workspace carve-out (256B aligned regions) ----
  char* ws = (char*)d_ws;
  size_t off = 0;
  auto alloc = [&](size_t bytes) -> char* {
    off = (off + 255) & ~(size_t)255;
    char* p = ws + off;
    off += bytes;
    return p;
  };

  __bf16* h0bf  = (__bf16*)alloc((size_t)N * 64 * 2);
  __bf16* wt1l  = (__bf16*)alloc((size_t)256 * 64 * 2);
  __bf16* wt1r  = (__bf16*)alloc((size_t)256 * 64 * 2);
  __bf16* wt2l  = (__bf16*)alloc((size_t)256 * 256 * 2);
  __bf16* wt2r  = (__bf16*)alloc((size_t)256 * 256 * 2);
  __bf16* h1bf  = (__bf16*)alloc((size_t)N * C * 2);
  float*  xl    = (float*)alloc((size_t)N * C * 4);
  float*  xr    = (float*)alloc((size_t)N * C * 4);
  float*  agg   = (float*)alloc((size_t)N * C * 4);
  float*  logit = (float*)alloc((size_t)E2 * H * 4);
  float*  lmax  = (float*)alloc((size_t)N * H * 4);
  float*  den   = (float*)alloc((size_t)N * H * 4);
  float*  pooled= (float*)alloc((size_t)B * C * 4);
  float*  part  = (float*)alloc(256 * 4);
  float*  eamean= (float*)alloc(4);
  int*    gs    = (int*)alloc(B * 4);
  int*    ge    = (int*)alloc(B * 4);

  const int TPB = 256;
  const int gNH  = (N * H + TPB - 1) / TPB;
  const int gNC  = (N * C + TPB - 1) / TPB;
  const int gEH  = (E2 * H + TPB - 1) / TPB;
  const int gE64 = (E2 * 64 + TPB - 1) / TPB;

  // ---- prologue: edge-attr mean, encoder, bf16 weight transposes ----
  k_ea_part<<<256, 256, 0, stream>>>(ea, part);
  k_ea_fin<<<1, 256, 0, stream>>>(part, eamean);
  k_encoder<<<(N * 64 + TPB - 1) / TPB, TPB, 0, stream>>>(x, enc_w, enc_b, h0bf);
  k_twt<<<(64 * 256 + TPB - 1) / TPB, TPB, 0, stream>>>(g1_wl, wt1l, 64);
  k_twt<<<(64 * 256 + TPB - 1) / TPB, TPB, 0, stream>>>(g1_wr, wt1r, 64);
  k_twt<<<(256 * 256 + TPB - 1) / TPB, TPB, 0, stream>>>(g2_wl, wt2l, 256);
  k_twt<<<(256 * 256 + TPB - 1) / TPB, TPB, 0, stream>>>(g2_wr, wt2r, 256);

  // ---- GATv2 layer 1 (Fin = 64) ----
  k_fill_f32<<<gNH, TPB, 0, stream>>>(lmax, -INFINITY, N * H);
  k_fill_f32<<<gNH, TPB, 0, stream>>>(den, 0.0f, N * H);
  k_fill_f32<<<gNC, TPB, 0, stream>>>(agg, 0.0f, N * C);
  k_gemm_dual<64><<<N / 16, 256, 0, stream>>>(h0bf, wt1l, wt1r, g1_bl, g1_br, xl, xr);
  k_edge_logits<<<gEH, TPB, 0, stream>>>(ei, ea, eamean, xl, xr, g1_we, g1_att, logit, lmax);
  k_edge_exp<<<gEH, TPB, 0, stream>>>(ei, logit, lmax, den);
  k_edge_agg<<<gE64, TPB, 0, stream>>>(ei, logit, den, xl, agg);
  k_postnode<<<gNC, TPB, 0, stream>>>(agg, g1_bias, nullptr, h1bf);  // bf16 h1

  // ---- GATv2 layer 2 (Fin = 256) ----
  k_fill_f32<<<gNH, TPB, 0, stream>>>(lmax, -INFINITY, N * H);
  k_fill_f32<<<gNH, TPB, 0, stream>>>(den, 0.0f, N * H);
  k_fill_f32<<<gNC, TPB, 0, stream>>>(agg, 0.0f, N * C);
  k_gemm_dual<256><<<N / 16, 256, 0, stream>>>(h1bf, wt2l, wt2r, g2_bl, g2_br, xl, xr);
  k_edge_logits<<<gEH, TPB, 0, stream>>>(ei, ea, eamean, xl, xr, g2_we, g2_att, logit, lmax);
  k_edge_exp<<<gEH, TPB, 0, stream>>>(ei, logit, lmax, den);
  k_edge_agg<<<gE64, TPB, 0, stream>>>(ei, logit, den, xl, agg);
  k_postnode<<<gNC, TPB, 0, stream>>>(agg, g2_bias, agg, nullptr);   // f32 h2 in place

  // ---- global mean pool (deterministic; batch sorted) + MLP head ----
  k_bounds_init<<<1, 64, 0, stream>>>(gs, ge);
  k_bounds<<<(N + TPB - 1) / TPB, TPB, 0, stream>>>(batch, gs, ge);
  k_pool<<<B, 256, 0, stream>>>(agg, gs, ge, pooled);
  k_head<<<B, 128, 0, stream>>>(pooled, p1_w, p1_b, ln_g, ln_b, p2_w, p2_b,
                                hc_w, hc_b, hf_w, hf_b, out);
}
